// DecoderLayer_59691455480067
// MI455X (gfx1250) — matrix operand order repaired and checked by reading, not compiled, not run on hardware
//
#include <hip/hip_runtime.h>
#include <hip/hip_bf16.h>

// MI455X (gfx1250) decoder layer. All GEMMs + attention on v_wmma_f32_16x16x32_f16,
// GEMM A/B tiles staged through LDS with double-buffered global_load_async_to_lds_b128.
// B=2, S=2048, D=1024, H=16, DK=64, FF=4096. fp32 in/out, f16 WMMA internally.

typedef __attribute__((ext_vector_type(16))) _Float16 v16h;
typedef __attribute__((ext_vector_type(8)))  float    v8f;
typedef __attribute__((ext_vector_type(4)))  unsigned int v4u;

union F16Frag {
    v16h v;
    unsigned int u[8];
    _Float16 h[16];
};

// compile-time model dims (all powers of two -> shifts, no integer division)
constexpr int Bc  = 2;
constexpr int Sc  = 2048;   // log2 = 11
constexpr int Dc  = 1024;
constexpr int Hc  = 16;
constexpr int DKc = 64;     // log2 = 6
constexpr int FFc = 4096;
constexpr int Mc  = Bc * Sc;  // 4096 tokens

// ---------------------------------------------------------------------------
// elementwise fp32 -> f16
__global__ __launch_bounds__(256)
void cvt_f16_kernel(const float* __restrict__ in, _Float16* __restrict__ out, long n) {
    long i = (long)blockIdx.x * blockDim.x + threadIdx.x;
    if (i < n) out[i] = (_Float16)in[i];
}

// fp32 [K,N] -> f16 [N,K]; K is a power of two (KSHIFT = log2 K)
template <int KSHIFT>
__global__ __launch_bounds__(256)
void cvt_f16_t_kernel(const float* __restrict__ w, _Float16* __restrict__ wt, int N) {
    constexpr int K = 1 << KSHIFT;
    long i = (long)blockIdx.x * blockDim.x + threadIdx.x;
    if (i < ((long)N << KSHIFT)) {
        int n = (int)(i >> KSHIFT);
        int k = (int)(i & (K - 1));
        wt[i] = (_Float16)w[(long)k * N + n];
    }
}

// ---------------------------------------------------------------------------
// WMMA GEMM: C[M,N] = A[M,K](f16,row-major) * Bt[N,K](f16)^T + bias.
// block = 256 threads (8 waves); block tile 128x128; wave tile 32x64 (2x4 WMMA).
// A and Bt tiles (128x32 f16 each) double-buffered through LDS via
// global_load_async_to_lds_b128; rows padded to 80B to spread banks.
// MODE (f16 out): 0 row-major [M,N]; 1 per-head [B,H,S,DK]; 2 per-head [B,H,DK,S].
constexpr int ROWPAD  = 40;             // f16 elements per staged row (32 data + 8 pad)
constexpr int TILEELT = 128 * ROWPAD;   // 5120 f16 per tile region
constexpr int BUFELT  = 2 * TILEELT;    // A region + B region per buffer
constexpr int GEMM_LDS_BYTES = 2 * BUFELT * 2;  // 40960 bytes

template <int MODE, int RELU>
__global__ __launch_bounds__(256)
void gemm_wmma_kernel(const _Float16* __restrict__ A, const _Float16* __restrict__ Bt,
                      const float* __restrict__ bias,
                      float* __restrict__ outF32, _Float16* __restrict__ outF16,
                      int N, int K) {
    extern __shared__ _Float16 smem[];   // dynamic-only => LDS offset base 0

    const int tid  = threadIdx.x;
    const int lane = tid & 31;
    const int wave = tid >> 5;
    const int wr   = wave >> 1;          // 0..3
    const int wc   = wave & 1;           // 0..1
    const int rowBase = blockIdx.y * 128;
    const int colBase = blockIdx.x * 128;
    const int half = lane >> 4;
    const int l16  = lane & 15;

    // stage tile (128 rows x 32 k f16) for A and Bt into LDS buffer `buf`
    auto issue_tile = [&](int kk, int buf) {
        const unsigned bufByte = (unsigned)buf * (BUFELT * 2);
        #pragma unroll
        for (int j = 0; j < 2; ++j) {
            int c   = tid + j * 256;          // chunk 0..511 (16B each)
            int row = c >> 2;
            int seg = c & 3;
            const _Float16* ga = A + (long)(rowBase + row) * K + kk + seg * 8;
            unsigned la = bufByte + (unsigned)(row * 80 + seg * 16);
            asm volatile("global_load_async_to_lds_b128 %0, %1, off"
                         :: "v"(la), "v"((unsigned long long)(const void*)ga)
                         : "memory");
            const _Float16* gb = Bt + (long)(colBase + row) * K + kk + seg * 8;
            unsigned lb = bufByte + (unsigned)(TILEELT * 2) + (unsigned)(row * 80 + seg * 16);
            asm volatile("global_load_async_to_lds_b128 %0, %1, off"
                         :: "v"(lb), "v"((unsigned long long)(const void*)gb)
                         : "memory");
        }
    };

    v8f acc[2][4];
    #pragma unroll
    for (int r = 0; r < 2; ++r)
        #pragma unroll
        for (int c = 0; c < 4; ++c)
            acc[r][c] = (v8f){0.f,0.f,0.f,0.f,0.f,0.f,0.f,0.f};

    const int nk = K >> 5;
    issue_tile(0, 0);

    for (int ik = 0; ik < nk; ++ik) {
        const int buf = ik & 1;
        asm volatile("s_wait_asynccnt 0x0" ::: "memory");
        __syncthreads();
        if (ik + 1 < nk) issue_tile((ik + 1) << 5, buf ^ 1);

        const _Float16* sA = smem + buf * BUFELT;
        const _Float16* sB = sA + TILEELT;

        F16Frag aF[2], bF[4];
        #pragma unroll
        for (int r = 0; r < 2; ++r) {
            const _Float16* p = sA + (wr * 32 + r * 16 + l16) * ROWPAD + half * 8;
            *(v4u*)&aF[r].u[0] = *(const v4u*)p;          // k = half*8 + 0..7
            *(v4u*)&aF[r].u[4] = *(const v4u*)(p + 16);   // k = 16 + half*8 + 0..7
        }
        #pragma unroll
        for (int c = 0; c < 4; ++c) {
            const _Float16* p = sB + (wc * 64 + c * 16 + l16) * ROWPAD + half * 8;
            *(v4u*)&bF[c].u[0] = *(const v4u*)p;          // k = half*8 + 0..7
            *(v4u*)&bF[c].u[4] = *(const v4u*)(p + 16);   // k = 16 + half*8 + 0..7
        }
        #pragma unroll
        for (int r = 0; r < 2; ++r)
            #pragma unroll
            for (int c = 0; c < 4; ++c)
                acc[r][c] = __builtin_amdgcn_wmma_f32_16x16x32_f16(
                    false, aF[r].v, false, bF[c].v, (short)0, acc[r][c], false, false);
    }

    // epilogue: C layout = col n in lane (l16), row m = v + 8*half per VGPR v
    #pragma unroll
    for (int r = 0; r < 2; ++r) {
        #pragma unroll
        for (int c = 0; c < 4; ++c) {
            int col = colBase + wc * 64 + c * 16 + l16;
            float bv = bias ? bias[col] : 0.f;
            #pragma unroll
            for (int v = 0; v < 8; ++v) {
                int row = rowBase + wr * 32 + r * 16 + v + 8 * half;
                float val = acc[r][c][v] + bv;
                if (RELU) val = fmaxf(val, 0.f);
                if (outF32) outF32[(long)row * N + col] = val;
                if (outF16) {
                    long idx;
                    if constexpr (MODE == 0) {
                        idx = (long)row * N + col;
                    } else {
                        int b = row >> 11, s = row & (Sc - 1);     // S = 2048
                        int h = col >> 6,  dk = col & (DKc - 1);   // DK = 64
                        if constexpr (MODE == 1)
                            idx = ((long)(b * Hc + h) * Sc + s) * DKc + dk;  // [B,H,S,DK]
                        else
                            idx = ((long)(b * Hc + h) * DKc + dk) * Sc + s;  // [B,H,DK,S]
                    }
                    outF16[idx] = (_Float16)val;
                }
            }
        }
    }
}

// ---------------------------------------------------------------------------
// Flash attention, online softmax, no 1/sqrt(dk) scale (matches reference).
// Computes S^T = K x Q^T: softmax is lane-local and the S^T C-register layout
// maps element-for-element onto the P A-fragment layout (zero cross-lane packing).
// grid = (S/128, B*H); 8 waves/block; each wave owns 16 queries.
template <int CAUSAL>
__global__ __launch_bounds__(256)
void attn_wmma_kernel(const _Float16* __restrict__ Qh,   // [B,H,S,DK]
                      const _Float16* __restrict__ Kh,   // [B,H,S,DK]
                      const _Float16* __restrict__ Vt,   // [B,H,DK,S]
                      _Float16* __restrict__ O) {        // [B*S, D] f16
    const int lane = threadIdx.x & 31;
    const int wave = threadIdx.x >> 5;
    const int half = lane >> 4;
    const int l16  = lane & 15;
    const int bh   = blockIdx.y;            // b*H + h
    const int b    = bh >> 4;               // H = 16
    const int h    = bh & (Hc - 1);
    const int qb   = blockIdx.x * 128 + wave * 16;

    const _Float16* Qp = Qh + (long)bh * Sc * DKc;
    const _Float16* Kp = Kh + (long)bh * Sc * DKc;
    const _Float16* Vp = Vt + (long)bh * DKc * Sc;

    // Q as B-matrix fragments (dk x queries), two 32-dk chunks
    F16Frag qf[2];
    #pragma unroll
    for (int ch = 0; ch < 2; ++ch) {
        const _Float16* qp = Qp + (long)(qb + l16) * DKc + ch * 32 + half * 8;
        #pragma unroll
        for (int i = 0; i < 8; ++i)
            qf[ch].u[i] = *(const unsigned int*)(qp + ((i < 4) ? 2 * i : 16 + 2 * (i - 4)));
    }

    v8f oacc[4];
    #pragma unroll
    for (int c = 0; c < 4; ++c)
        oacc[c] = (v8f){0.f,0.f,0.f,0.f,0.f,0.f,0.f,0.f};

    float m_run = -1e30f, l_run = 0.f;
    const int kend = CAUSAL ? (qb + 16) : Sc;

    for (int kb = 0; kb < kend; kb += 32) {
        // scores: S^T tiles (rows = keys, cols = queries)
        v8f sa[2];
        #pragma unroll
        for (int t = 0; t < 2; ++t) {
            sa[t] = (v8f){0.f,0.f,0.f,0.f,0.f,0.f,0.f,0.f};
            #pragma unroll
            for (int ch = 0; ch < 2; ++ch) {
                F16Frag af;
                const _Float16* kp = Kp + (long)(kb + t * 16 + l16) * DKc + ch * 32 + half * 8;
                #pragma unroll
                for (int i = 0; i < 8; ++i) {
                    int off = (i < 4) ? 2 * i : 16 + 2 * (i - 4);
                    af.u[i] = *(const unsigned int*)(kp + off);
                }
                sa[t] = __builtin_amdgcn_wmma_f32_16x16x32_f16(
                    false, af.v, false, qf[ch].v, (short)0, sa[t], false, false);
            }
        }

        // lane-local softmax over this lane's 16 keys; merge halves with one shfl
        const int q = qb + l16;
        float p0[8], p1[8];
        float lm = -1e30f;
        #pragma unroll
        for (int v = 0; v < 8; ++v) {
            float s0 = sa[0][v], s1 = sa[1][v];
            if (CAUSAL) {
                if (kb + v + 8 * half > q)      s0 = -1e30f;
                if (kb + 16 + v + 8 * half > q) s1 = -1e30f;
            }
            p0[v] = s0; p1[v] = s1;
            lm = fmaxf(lm, fmaxf(s0, s1));
        }
        lm = fmaxf(lm, __shfl_xor(lm, 16, 32));
        float m_new = fmaxf(m_run, lm);
        float scale = __expf(m_run - m_new);

        // P packs directly into the A-fragment layout
        F16Frag pa;
        float ps = 0.f;
        #pragma unroll
        for (int v = 0; v < 8; ++v) {
            float e0 = __expf(p0[v] - m_new);
            float e1 = __expf(p1[v] - m_new);
            ps += e0 + e1;
            pa.h[v]     = (_Float16)e0;
            pa.h[v + 8] = (_Float16)e1;
        }
        ps += __shfl_xor(ps, 16, 32);
        l_run = l_run * scale + ps;
        m_run = m_new;

        // broadcast per-row rescale factors once (rows m = v + 8*half)
        float f[8];
        #pragma unroll
        for (int v = 0; v < 8; ++v) f[v] = __shfl(scale, v + 8 * half, 32);
        #pragma unroll
        for (int c = 0; c < 4; ++c)
            #pragma unroll
            for (int v = 0; v < 8; ++v)
                oacc[c][v] *= f[v];

        // out += P x V; V^T layout makes B-fragment loads contiguous
        #pragma unroll
        for (int c = 0; c < 4; ++c) {
            F16Frag vf;
            const _Float16* vp = Vp + (long)(c * 16 + l16) * Sc + kb + half * 8;
            #pragma unroll
            for (int i = 0; i < 8; ++i)
                vf.u[i] = *(const unsigned int*)(vp + ((i < 4) ? 2 * i : 16 + 2 * (i - 4)));
            oacc[c] = __builtin_amdgcn_wmma_f32_16x16x32_f16(
                false, pa.v, false, vf.v, (short)0, oacc[c], false, false);
        }
    }

    // normalize and scatter to [token, D] f16
    float linv[8];
    #pragma unroll
    for (int v = 0; v < 8; ++v) linv[v] = 1.f / __shfl(l_run, v + 8 * half, 32);
    #pragma unroll
    for (int c = 0; c < 4; ++c) {
        int dk = c * 16 + l16;
        #pragma unroll
        for (int v = 0; v < 8; ++v) {
            int qq = qb + v + 8 * half;
            O[(long)(b * Sc + qq) * Dc + h * DKc + dk] = (_Float16)(oacc[c][v] * linv[v]);
        }
    }
}

// ---------------------------------------------------------------------------
// LayerNorm( A + R ) -> Y (fp32) and optional Yh (f16). One block per row, D=1024.
__global__ __launch_bounds__(256)
void ln_kernel(const float* __restrict__ A, const float* __restrict__ R,
               const float* __restrict__ g, const float* __restrict__ be,
               float* __restrict__ Y, _Float16* __restrict__ Yh) {
    __shared__ float red[8];
    const int row = blockIdx.x;
    const float* a = A + (long)row * Dc;
    const float* r = R ? R + (long)row * Dc : nullptr;

    float v[4];
    float sum = 0.f;
    #pragma unroll
    for (int i = 0; i < 4; ++i) {
        int idx = threadIdx.x + i * 256;
        float x = a[idx] + (r ? r[idx] : 0.f);
        v[i] = x; sum += x;
    }
    #pragma unroll
    for (int o = 16; o > 0; o >>= 1) sum += __shfl_xor(sum, o, 32);
    if ((threadIdx.x & 31) == 0) red[threadIdx.x >> 5] = sum;
    __syncthreads();
    float tot = 0.f;
    #pragma unroll
    for (int i = 0; i < 8; ++i) tot += red[i];
    float mean = tot * (1.f / Dc);
    __syncthreads();

    float vs = 0.f;
    #pragma unroll
    for (int i = 0; i < 4; ++i) { float d = v[i] - mean; vs += d * d; }
    #pragma unroll
    for (int o = 16; o > 0; o >>= 1) vs += __shfl_xor(vs, o, 32);
    if ((threadIdx.x & 31) == 0) red[threadIdx.x >> 5] = vs;
    __syncthreads();
    float tv = 0.f;
    #pragma unroll
    for (int i = 0; i < 8; ++i) tv += red[i];
    float rstd = rsqrtf(tv * (1.f / Dc) + 1e-5f);

    #pragma unroll
    for (int i = 0; i < 4; ++i) {
        int idx = threadIdx.x + i * 256;
        float y = (v[i] - mean) * rstd * g[idx] + be[idx];
        Y[(long)row * Dc + idx] = y;
        if (Yh) Yh[(long)row * Dc + idx] = (_Float16)y;
    }
}

// ---------------------------------------------------------------------------
extern "C" void kernel_launch(void* const* d_in, const int* in_sizes, int n_in,
                              void* d_out, int out_size, void* d_ws, size_t ws_size,
                              hipStream_t stream) {
    const float* x   = (const float*)d_in[0];
    const float* enc = (const float*)d_in[1];
    // d_in[2]/d_in[3] masks: causal handled arithmetically, src mask is all-ones.
    const float* wq1 = (const float*)d_in[4];  const float* bq1 = (const float*)d_in[5];
    const float* wk1 = (const float*)d_in[6];  const float* bk1 = (const float*)d_in[7];
    const float* wv1 = (const float*)d_in[8];  const float* bv1 = (const float*)d_in[9];
    const float* wo1 = (const float*)d_in[10]; const float* bo1 = (const float*)d_in[11];
    const float* g1  = (const float*)d_in[12]; const float* be1 = (const float*)d_in[13];
    const float* wq2 = (const float*)d_in[14]; const float* bq2 = (const float*)d_in[15];
    const float* wk2 = (const float*)d_in[16]; const float* bk2 = (const float*)d_in[17];
    const float* wv2 = (const float*)d_in[18]; const float* bv2 = (const float*)d_in[19];
    const float* wo2 = (const float*)d_in[20]; const float* bo2 = (const float*)d_in[21];
    const float* g2  = (const float*)d_in[22]; const float* be2 = (const float*)d_in[23];
    const float* fw1 = (const float*)d_in[24]; const float* fb1 = (const float*)d_in[25];
    const float* fw2 = (const float*)d_in[26]; const float* fb2 = (const float*)d_in[27];
    const float* g3  = (const float*)d_in[28]; const float* be3 = (const float*)d_in[29];

    // bump allocator over d_ws
    size_t off = 0;
    char* base = (char*)d_ws;
    auto alloc = [&](size_t bytes) -> void* {
        void* p = base + off;
        off += (bytes + 255) & ~(size_t)255;
        return p;
    };
    auto h2 = [&](size_t n) { return (_Float16*)alloc(n * sizeof(_Float16)); };
    auto f4 = [&](size_t n) { return (float*)alloc(n * sizeof(float)); };

    _Float16 *wq1t = h2((size_t)Dc*Dc), *wk1t = h2((size_t)Dc*Dc), *wv1t = h2((size_t)Dc*Dc), *wo1t = h2((size_t)Dc*Dc);
    _Float16 *wq2t = h2((size_t)Dc*Dc), *wk2t = h2((size_t)Dc*Dc), *wv2t = h2((size_t)Dc*Dc), *wo2t = h2((size_t)Dc*Dc);
    _Float16 *w1t  = h2((size_t)Dc*FFc), *w2t = h2((size_t)FFc*Dc);
    _Float16 *xh   = h2((size_t)Mc*Dc),  *ench = h2((size_t)Mc*Dc);
    _Float16 *Q1 = h2((size_t)Mc*Dc), *K1 = h2((size_t)Mc*Dc), *V1t = h2((size_t)Mc*Dc);
    _Float16 *Q2 = h2((size_t)Mc*Dc), *K2 = h2((size_t)Mc*Dc), *V2t = h2((size_t)Mc*Dc);
    _Float16 *O1 = h2((size_t)Mc*Dc), *O2 = h2((size_t)Mc*Dc);
    _Float16 *x2h = h2((size_t)Mc*Dc), *x4h = h2((size_t)Mc*Dc);
    _Float16 *ffh = h2((size_t)Mc*FFc);
    float *x1 = f4((size_t)Mc*Dc), *x2 = f4((size_t)Mc*Dc), *x3 = f4((size_t)Mc*Dc);
    float *x4 = f4((size_t)Mc*Dc), *x5 = f4((size_t)Mc*Dc);

    const dim3 blk(256);

    // --- precision prep ---
    {
        long n = (long)Mc * Dc;
        cvt_f16_kernel<<<dim3((unsigned)((n + 255) / 256)), blk, 0, stream>>>(x, xh, n);
        cvt_f16_kernel<<<dim3((unsigned)((n + 255) / 256)), blk, 0, stream>>>(enc, ench, n);
        unsigned gw = (unsigned)(((long)Dc * Dc + 255) / 256);
        cvt_f16_t_kernel<10><<<gw, blk, 0, stream>>>(wq1, wq1t, Dc);
        cvt_f16_t_kernel<10><<<gw, blk, 0, stream>>>(wk1, wk1t, Dc);
        cvt_f16_t_kernel<10><<<gw, blk, 0, stream>>>(wv1, wv1t, Dc);
        cvt_f16_t_kernel<10><<<gw, blk, 0, stream>>>(wo1, wo1t, Dc);
        cvt_f16_t_kernel<10><<<gw, blk, 0, stream>>>(wq2, wq2t, Dc);
        cvt_f16_t_kernel<10><<<gw, blk, 0, stream>>>(wk2, wk2t, Dc);
        cvt_f16_t_kernel<10><<<gw, blk, 0, stream>>>(wv2, wv2t, Dc);
        cvt_f16_t_kernel<10><<<gw, blk, 0, stream>>>(wo2, wo2t, Dc);
        unsigned gf = (unsigned)(((long)Dc * FFc + 255) / 256);
        cvt_f16_t_kernel<10><<<gf, blk, 0, stream>>>(fw1, w1t, FFc);  // [FF][D],  K=D=1024
        cvt_f16_t_kernel<12><<<gf, blk, 0, stream>>>(fw2, w2t, Dc);   // [D][FF],  K=FF=4096
    }

    auto gemm = [&](auto kfn, const _Float16* A_, const _Float16* Bt_, const float* bias_,
                    float* o32, _Float16* o16, int Mm, int Nn, int Kk) {
        dim3 grid(Nn / 128, Mm / 128);
        kfn<<<grid, blk, GEMM_LDS_BYTES, stream>>>(A_, Bt_, bias_, o32, o16, Nn, Kk);
    };
    const dim3 agrid(Sc / 128, Bc * Hc);

    // --- self attention ---
    gemm(gemm_wmma_kernel<1,0>, xh, wq1t, bq1, nullptr, Q1,  Mc, Dc, Dc);
    gemm(gemm_wmma_kernel<1,0>, xh, wk1t, bk1, nullptr, K1,  Mc, Dc, Dc);
    gemm(gemm_wmma_kernel<2,0>, xh, wv1t, bv1, nullptr, V1t, Mc, Dc, Dc);
    attn_wmma_kernel<1><<<agrid, blk, 0, stream>>>(Q1, K1, V1t, O1);
    gemm(gemm_wmma_kernel<0,0>, O1, wo1t, bo1, x1, nullptr, Mc, Dc, Dc);
    ln_kernel<<<Mc, blk, 0, stream>>>(x, x1, g1, be1, x2, x2h);

    // --- cross attention ---
    gemm(gemm_wmma_kernel<1,0>, x2h,  wq2t, bq2, nullptr, Q2,  Mc, Dc, Dc);
    gemm(gemm_wmma_kernel<1,0>, ench, wk2t, bk2, nullptr, K2,  Mc, Dc, Dc);
    gemm(gemm_wmma_kernel<2,0>, ench, wv2t, bv2, nullptr, V2t, Mc, Dc, Dc);
    attn_wmma_kernel<0><<<agrid, blk, 0, stream>>>(Q2, K2, V2t, O2);
    gemm(gemm_wmma_kernel<0,0>, O2, wo2t, bo2, x3, nullptr, Mc, Dc, Dc);
    ln_kernel<<<Mc, blk, 0, stream>>>(x2, x3, g2, be2, x4, x4h);

    // --- feed forward ---
    gemm(gemm_wmma_kernel<0,1>, x4h, w1t, fb1, nullptr, ffh, Mc, FFc, Dc);  // ReLU
    gemm(gemm_wmma_kernel<0,0>, ffh, w2t, fb2, x5, nullptr, Mc, Dc, FFc);
    ln_kernel<<<Mc, blk, 0, stream>>>(x4, x5, g3, be3, (float*)d_out, nullptr);

    (void)in_sizes; (void)n_in; (void)out_size; (void)ws_size;
}
